// TracedDecoderStep_79405355368945
// MI455X (gfx1250) — compile-verified, matrix operand order checked by reading, not run on hardware
//
#include <hip/hip_runtime.h>
#include <math.h>

// ---------------------------------------------------------------------------
// Tacotron2 GMM-attention decoder step for MI455X (gfx1250, wave32, WMMA)
// Memory-bound (~205 MB/call -> ~9us @ 23.3 TB/s). GEMMs on v_wmma_f32_16x16x32_f16
// with K-split for WGP-level bandwidth parallelism; transcendental stages on VALU;
// intermediates L2-resident.
// ---------------------------------------------------------------------------

typedef __attribute__((ext_vector_type(16))) _Float16 v16h;
typedef __attribute__((ext_vector_type(8)))  float    v8f;

#define BATCH   64
#define NMEL    80
#define FENC    512
#define PRENETD 256
#define ARNN    1024
#define DRNN    1024
#define TTLEN   1000
#define KMIX    8
#define NGATES  4096   // 4*1024

// ---------------------------------------------------------------------------
// Fused LSTM-gate GEMM (K-split partial):
//   gates_part[kz][64][4096] = [s0 | s1 | h][:, kz-slice] @ W[:, kz-slice]^T
// A = activations (64 x Kchunk) staged in LDS as f16; each 32-wide K-chunk lies
// entirely inside one segment (all boundaries are multiples of 32), so segment
// selection is uniform scalar work per chunk. One wave per 16-col N-tile,
// 4 M-tiles (full batch). Biases are applied later in the pointwise kernel.
// ---------------------------------------------------------------------------
__launch_bounds__(256)
__global__ void lstm_gates_wmma(const float* __restrict__ s0, int len0,
                                const float* __restrict__ s1, int len1,
                                const float* __restrict__ hprev, int lenh,
                                const float* __restrict__ wih,
                                const float* __restrict__ whh,
                                int Kchunk,
                                float* __restrict__ gates_part)
{
    __shared__ alignas(16) _Float16 As[64][32];   // 64x32 f16 chunk = 4 KB

    const int tid   = threadIdx.x;
    const int wave  = tid >> 5;
    const int lane  = tid & 31;
    const int lrow  = lane & 15;       // M (A-frag) / N (B-frag) row within tile
    const int khalf = lane >> 4;       // which K half this lane holds
    const int ntile = blockIdx.x * 8 + wave;
    const int n     = ntile * 16 + lrow;        // output column (gate index)
    const int Kih   = len0 + len1;

    const int kz     = blockIdx.y;
    const int kbegin = kz * Kchunk;
    const int kend   = kbegin + Kchunk;

    // per-thread A-staging coordinates: 4 threads per row, 8 contiguous k each
    const int arow = tid >> 2;
    const int akk  = (tid & 3) * 8;

    v8f zero = {0.f,0.f,0.f,0.f,0.f,0.f,0.f,0.f};
    v8f acc[4];
    acc[0] = zero; acc[1] = zero; acc[2] = zero; acc[3] = zero;

    for (int k0 = kbegin; k0 < kend; k0 += 32) {
        // ---- uniform (scalar) segment select for this K-chunk ----
        const float* aseg; int astride, koff;
        if (k0 < len0)      { aseg = s0;    astride = len0; koff = k0; }
        else if (k0 < Kih)  { aseg = s1;    astride = len1; koff = k0 - len0; }
        else                { aseg = hprev; astride = lenh; koff = k0 - Kih; }

        __syncthreads();   // protect As against previous iteration readers

        // ---- stage A chunk: 2x float4 global load -> 8x f16 -> 1x ds_store_b128 ----
        {
            const float4* ap = (const float4*)(aseg + (size_t)arow * astride + koff + akk);
            float4 a0 = ap[0];
            float4 a1 = ap[1];
            union { uint4 q; _Float16 h[8]; } pk;
            pk.h[0]=(_Float16)a0.x; pk.h[1]=(_Float16)a0.y;
            pk.h[2]=(_Float16)a0.z; pk.h[3]=(_Float16)a0.w;
            pk.h[4]=(_Float16)a1.x; pk.h[5]=(_Float16)a1.y;
            pk.h[6]=(_Float16)a1.z; pk.h[7]=(_Float16)a1.w;
            *(uint4*)(&As[arow][akk]) = pk.q;
        }
        __syncthreads();

        // ---- B fragment: weight row n, this K-chunk (row-major N x K == B layout) ----
        const float* wrow;
        if (k0 < Kih) wrow = wih + (size_t)n * Kih  + k0;
        else          wrow = whh + (size_t)n * lenh + (k0 - Kih);
        __builtin_prefetch((const void*)(wrow + 32), 0, 1);   // next chunk -> global_prefetch_b8
        const float4* wp = (const float4*)wrow;               // 16B aligned
        float4 f0 = wp[2 * khalf];        // K = 8*khalf .. +3
        float4 f1 = wp[2 * khalf + 1];    // K = 8*khalf+4 .. +7
        float4 f2 = wp[4 + 2 * khalf];    // K = 16+8*khalf .. +3
        float4 f3 = wp[4 + 2 * khalf + 1];
        v16h bfrag;
        bfrag[0]=(_Float16)f0.x;  bfrag[1]=(_Float16)f0.y;
        bfrag[2]=(_Float16)f0.z;  bfrag[3]=(_Float16)f0.w;
        bfrag[4]=(_Float16)f1.x;  bfrag[5]=(_Float16)f1.y;
        bfrag[6]=(_Float16)f1.z;  bfrag[7]=(_Float16)f1.w;
        bfrag[8]=(_Float16)f2.x;  bfrag[9]=(_Float16)f2.y;
        bfrag[10]=(_Float16)f2.z; bfrag[11]=(_Float16)f2.w;
        bfrag[12]=(_Float16)f3.x; bfrag[13]=(_Float16)f3.y;
        bfrag[14]=(_Float16)f3.z; bfrag[15]=(_Float16)f3.w;

        // ---- preload all 4 A fragments, then 4 back-to-back WMMAs ----
        union { v16h v; uint4 q[2]; } af[4];
#pragma unroll
        for (int mt = 0; mt < 4; ++mt) {
            const uint4* lp = (const uint4*)(&As[mt * 16 + lrow][0]);
            af[mt].q[0] = lp[khalf];        // K = 8*khalf..+7   -> elems 0..7
            af[mt].q[1] = lp[2 + khalf];    // K = 16+8*khalf..  -> elems 8..15
        }
#pragma unroll
        for (int mt = 0; mt < 4; ++mt) {
            acc[mt] = __builtin_amdgcn_wmma_f32_16x16x32_f16(
                          false, af[mt].v, false, bfrag, (short)0, acc[mt], false, false);
        }
    }

    // ---- epilogue: store partial. D layout: lane%16 = N, VGPR r -> M = r+8*khalf ----
    float* gp = gates_part + (size_t)kz * BATCH * NGATES;
#pragma unroll
    for (int mt = 0; mt < 4; ++mt) {
#pragma unroll
        for (int r = 0; r < 8; ++r) {
            int b = mt * 16 + r + 8 * khalf;
            gp[(size_t)b * NGATES + n] = acc[mt][r];
        }
    }
}

// ---------------------------------------------------------------------------
// LSTM pointwise: sum K-split partials + biases, then i,f,g,o -> h,c
// ---------------------------------------------------------------------------
__global__ void lstm_pointwise(const float* __restrict__ gates_part, int ksplit,
                               const float* __restrict__ bih,
                               const float* __restrict__ bhh,
                               const float* __restrict__ cprev,
                               float* __restrict__ hout,
                               float* __restrict__ cout)
{
    int idx = blockIdx.x * blockDim.x + threadIdx.x;
    if (idx >= BATCH * ARNN) return;
    int b = idx >> 10, j = idx & 1023;
    float gi = bih[j]        + bhh[j];
    float gf = bih[1024 + j] + bhh[1024 + j];
    float gg = bih[2048 + j] + bhh[2048 + j];
    float go = bih[3072 + j] + bhh[3072 + j];
    for (int p = 0; p < ksplit; ++p) {
        const float* g = gates_part + (size_t)p * BATCH * NGATES + (size_t)b * NGATES;
        gi += g[j]; gf += g[1024 + j]; gg += g[2048 + j]; go += g[3072 + j];
    }
    float si = 1.f / (1.f + __expf(-gi));
    float sf = 1.f / (1.f + __expf(-gf));
    float so = 1.f / (1.f + __expf(-go));
    float c2 = sf * cprev[idx] + si * tanhf(gg);
    float h2 = so * tanhf(c2);
    cout[idx] = c2;
    hout[idx] = h2;
}

// ---------------------------------------------------------------------------
// Prenet layer: out[b][j] = relu(in[b] . w[j]) * dmask[j]   (tiny)
// ---------------------------------------------------------------------------
__global__ void prenet_layer(const float* __restrict__ in, int Kin,
                             const float* __restrict__ w,
                             const float* __restrict__ dmask,
                             float* __restrict__ out)
{
    int idx = blockIdx.x * blockDim.x + threadIdx.x;
    if (idx >= BATCH * PRENETD) return;
    int b = idx >> 8, j = idx & 255;
    const float* wr = w + (size_t)j * Kin;
    const float* xr = in + (size_t)b * Kin;
    float s = 0.f;
    for (int k = 0; k < Kin; ++k) s += xr[k] * wr[k];
    out[idx] = fmaxf(s, 0.f) * dmask[j];
}

// ---------------------------------------------------------------------------
// GMM MLP: p[b][n] = [ah | ctx_prev] . mlp_w[n] + mlp_b[n]   (B=64, 3K=24)
// ---------------------------------------------------------------------------
__global__ void gmm_mlp(const float* __restrict__ ah,
                        const float* __restrict__ ctx,
                        const float* __restrict__ w,
                        const float* __restrict__ bias,
                        float* __restrict__ p)
{
    int b = blockIdx.x, n = threadIdx.x;
    if (n >= 3 * KMIX) return;
    const float* wr = w + (size_t)n * (ARNN + FENC);
    float s = bias[n];
    for (int k = 0; k < ARNN; ++k) s += ah[b * ARNN + k] * wr[k];
    for (int k = 0; k < FENC; ++k) s += ctx[b * FENC + k] * wr[ARNN + k];
    p[b * (3 * KMIX) + n] = s;
}

__device__ inline float softplusf(float x) {
    return (x > 20.f) ? x : log1pf(__expf(x));
}

// ---------------------------------------------------------------------------
// Mixture params: softmax(w_hat)/Z, softplus shifts/widths, new_mu
// ---------------------------------------------------------------------------
__global__ void gmm_params(const float* __restrict__ p,
                           const float* __restrict__ last_mu,
                           float* __restrict__ wz,
                           float* __restrict__ sigma_out,
                           float* __restrict__ new_mu)
{
    int b = threadIdx.x;
    if (b >= BATCH) return;
    const float* pb = p + b * 3 * KMIX;
    float wh[KMIX], m = -3.4e38f;
#pragma unroll
    for (int k = 0; k < KMIX; ++k) { wh[k] = pb[k]; m = fmaxf(m, wh[k]); }
    float sum = 0.f;
#pragma unroll
    for (int k = 0; k < KMIX; ++k) { wh[k] = __expf(wh[k] - m); sum += wh[k]; }
    float inv = 1.f / sum;
    const float SQRT2PI = 2.50662827463f;
#pragma unroll
    for (int k = 0; k < KMIX; ++k) {
        float w     = wh[k] * inv;
        float delta = softplusf(pb[KMIX + k]);
        float sg    = softplusf(pb[2 * KMIX + k]) + 1e-5f;
        wz[b * KMIX + k]        = w / (SQRT2PI * sg);
        sigma_out[b * KMIX + k] = sg;
        new_mu[b * KMIX + k]    = last_mu[b * KMIX + k] + delta;
    }
}

// ---------------------------------------------------------------------------
// Energies over Tt=1000 + masked softmax, one block per batch row.
// ---------------------------------------------------------------------------
__launch_bounds__(256)
__global__ void gmm_energy_softmax(const float* __restrict__ wz,
                                   const float* __restrict__ mu,
                                   const float* __restrict__ sigma,
                                   const unsigned char* __restrict__ mask,
                                   float* __restrict__ attn)
{
    int b = blockIdx.x, tid = threadIdx.x;
    __shared__ float e[1024];
    __shared__ float red[256];

    float lwz[KMIX], lmu[KMIX], linv[KMIX];
#pragma unroll
    for (int k = 0; k < KMIX; ++k) {
        lwz[k]  = wz[b * KMIX + k];
        lmu[k]  = mu[b * KMIX + k];
        linv[k] = 1.f / sigma[b * KMIX + k];
    }
    if (tid < 1024 - TTLEN) e[TTLEN + tid] = -3.4e38f;   // pad for reductions

    for (int t = tid; t < TTLEN; t += 256) {
        float en = 0.f;
#pragma unroll
        for (int k = 0; k < KMIX; ++k) {
            float z = ((float)t - lmu[k]) * linv[k];
            en += lwz[k] * __expf(-0.5f * z * z);
        }
        if (mask[b * TTLEN + t]) en = -1e9f;
        e[t] = en;
    }
    __syncthreads();

    // max reduce
    float m = fmaxf(fmaxf(e[tid], e[tid + 256]), fmaxf(e[tid + 512], e[tid + 768]));
    red[tid] = m; __syncthreads();
    for (int s = 128; s > 0; s >>= 1) {
        if (tid < s) red[tid] = fmaxf(red[tid], red[tid + s]);
        __syncthreads();
    }
    m = red[0];
    __syncthreads();

    // exp + sum reduce
    float ls = 0.f;
    for (int t = tid; t < TTLEN; t += 256) {
        float s_ = __expf(e[t] - m);
        e[t] = s_;
        ls += s_;
    }
    red[tid] = ls; __syncthreads();
    for (int s = 128; s > 0; s >>= 1) {
        if (tid < s) red[tid] += red[tid + s];
        __syncthreads();
    }
    float inv = 1.f / red[0];
    for (int t = tid; t < TTLEN; t += 256)
        attn[b * TTLEN + t] = e[t] * inv;
}

// ---------------------------------------------------------------------------
// Context partial: part[ts][b][f] = sum_{t in ts-slice} attn[b][t]*memory[b][t][f]
// grid = (FENC/256, BATCH, tsplit); attention slice cached in LDS; f-coalesced.
// ---------------------------------------------------------------------------
__launch_bounds__(256)
__global__ void context_partial(const float* __restrict__ attn,
                                const float* __restrict__ memory,
                                int tchunk,
                                float* __restrict__ part)
{
    int b  = blockIdx.y;
    int f  = blockIdx.x * 256 + threadIdx.x;
    int t0 = blockIdx.z * tchunk;
    int t1 = t0 + tchunk; if (t1 > TTLEN) t1 = TTLEN;
    int cnt = t1 - t0;

    __shared__ float aw[TTLEN];
    for (int t = threadIdx.x; t < cnt; t += 256) aw[t] = attn[b * TTLEN + t0 + t];
    __syncthreads();

    const float* mb = memory + ((size_t)b * TTLEN + t0) * FENC + f;
    float acc = 0.f;
    int t = 0;
    for (; t + 4 <= cnt; t += 4) {
        acc += aw[t]     * mb[(size_t)(t)     * FENC];
        acc += aw[t + 1] * mb[(size_t)(t + 1) * FENC];
        acc += aw[t + 2] * mb[(size_t)(t + 2) * FENC];
        acc += aw[t + 3] * mb[(size_t)(t + 3) * FENC];
    }
    for (; t < cnt; ++t) acc += aw[t] * mb[(size_t)t * FENC];
    part[((size_t)blockIdx.z * BATCH + b) * FENC + f] = acc;
}

__global__ void context_combine(const float* __restrict__ part, int tsplit,
                                float* __restrict__ ctx)
{
    int idx = blockIdx.x * blockDim.x + threadIdx.x;
    if (idx >= BATCH * FENC) return;
    float s = 0.f;
    for (int p = 0; p < tsplit; ++p) s += part[(size_t)p * BATCH * FENC + idx];
    ctx[idx] = s;
}

// ---------------------------------------------------------------------------
// Heads: decoder_output[b][0..79], gate[b] from [dh | ctx] (tiny)
// ---------------------------------------------------------------------------
__global__ void heads_kernel(const float* __restrict__ dh,
                             const float* __restrict__ ctx,
                             const float* __restrict__ proj_w,
                             const float* __restrict__ proj_b,
                             const float* __restrict__ gate_w,
                             const float* __restrict__ gate_b,
                             float* __restrict__ dec_out,
                             float* __restrict__ gate_out)
{
    int b = blockIdx.x, n = threadIdx.x;
    if (n > NMEL) return;
    const float* wr = (n < NMEL) ? (proj_w + (size_t)n * (DRNN + FENC)) : gate_w;
    float s = (n < NMEL) ? proj_b[n] : gate_b[0];
    for (int k = 0; k < DRNN; ++k) s += dh[b * DRNN + k] * wr[k];
    for (int k = 0; k < FENC; ++k) s += ctx[b * FENC + k] * wr[DRNN + k];
    if (n < NMEL) dec_out[b * NMEL + n] = s;
    else          gate_out[b] = s;
}

// ---------------------------------------------------------------------------
// Launcher
// ---------------------------------------------------------------------------
static inline size_t ws_need_bytes(int ksplit, int tsplit)
{
    size_t fl = 2u * BATCH * PRENETD                 // prenet x1, x
              + (size_t)ksplit * BATCH * NGATES      // gate partials
              + BATCH * 3 * KMIX + 2u * BATCH * KMIX // mlp p, wz, sigma
              + (size_t)tsplit * BATCH * FENC;       // ctx partials
    return fl * sizeof(float);
}

extern "C" void kernel_launch(void* const* d_in, const int* in_sizes, int n_in,
                              void* d_out, int out_size, void* d_ws, size_t ws_size,
                              hipStream_t stream)
{
    (void)in_sizes; (void)n_in; (void)out_size;

    const float* decoder_input     = (const float*)d_in[0];
    const float* attention_hidden  = (const float*)d_in[1];
    const float* attention_cell    = (const float*)d_in[2];
    const float* decoder_hidden    = (const float*)d_in[3];
    const float* decoder_cell      = (const float*)d_in[4];
    const float* attention_context = (const float*)d_in[5];
    const float* memory            = (const float*)d_in[6];
    const unsigned char* mask      = (const unsigned char*)d_in[7];
    const float* last_mu           = (const float*)d_in[8];
    const float* prenet_w1         = (const float*)d_in[9];
    const float* prenet_w2         = (const float*)d_in[10];
    const float* prenet_m1         = (const float*)d_in[11];
    const float* prenet_m2         = (const float*)d_in[12];
    const float* attn_wih          = (const float*)d_in[13];
    const float* attn_whh          = (const float*)d_in[14];
    const float* attn_bih          = (const float*)d_in[15];
    const float* attn_bhh          = (const float*)d_in[16];
    const float* dec_wih           = (const float*)d_in[17];
    const float* dec_whh           = (const float*)d_in[18];
    const float* dec_bih           = (const float*)d_in[19];
    const float* dec_bhh           = (const float*)d_in[20];
    const float* mlp_w             = (const float*)d_in[21];
    const float* mlp_b             = (const float*)d_in[22];
    const float* proj_w            = (const float*)d_in[23];
    const float* proj_b            = (const float*)d_in[24];
    const float* gate_w            = (const float*)d_in[25];
    const float* gate_b            = (const float*)d_in[26];

    // deterministic split selection from ws_size (constant across calls)
    int ksplit = 4, tsplit = 5;
    if (ws_size < ws_need_bytes(4, 5)) { ksplit = 2; tsplit = 2; }
    if (ws_size < ws_need_bytes(2, 2)) { ksplit = 1; tsplit = 1; }
    const int tchunk = (TTLEN + tsplit - 1) / tsplit;

    float* out = (float*)d_out;
    // output layout (flat, reference return order)
    float* o_decoder = out;                         // 64*80
    float* o_gate    = o_decoder + BATCH * NMEL;    // 64
    float* o_ah      = o_gate + BATCH;              // 64*1024
    float* o_ac      = o_ah + BATCH * ARNN;         // 64*1024
    float* o_dh      = o_ac + BATCH * ARNN;         // 64*1024
    float* o_dc      = o_dh + BATCH * DRNN;         // 64*1024
    float* o_attn    = o_dc + BATCH * DRNN;         // 64*1000
    float* o_ctx     = o_attn + BATCH * TTLEN;      // 64*512
    float* o_mu      = o_ctx + BATCH * FENC;        // 64*8

    // workspace layout
    float* ws      = (float*)d_ws;
    float* ws_x1   = ws;                                    // 64*256
    float* ws_x    = ws_x1 + BATCH * PRENETD;               // 64*256
    float* ws_gate = ws_x + BATCH * PRENETD;                // ksplit * 64*4096
    float* ws_p    = ws_gate + (size_t)ksplit * BATCH * NGATES; // 64*24
    float* ws_wz   = ws_p + BATCH * 3 * KMIX;               // 64*8
    float* ws_sg   = ws_wz + BATCH * KMIX;                  // 64*8
    float* ws_ctxp = ws_sg + BATCH * KMIX;                  // tsplit * 64*512

    // 1) prenet
    prenet_layer<<<64, 256, 0, stream>>>(decoder_input, NMEL,    prenet_w1, prenet_m1, ws_x1);
    prenet_layer<<<64, 256, 0, stream>>>(ws_x1,         PRENETD, prenet_w2, prenet_m2, ws_x);

    // 2) attention LSTM: [x(256) | ctx_prev(512)] + h(1024), K=1792
    lstm_gates_wmma<<<dim3(32, ksplit), 256, 0, stream>>>(
        ws_x, PRENETD, attention_context, FENC, attention_hidden, ARNN,
        attn_wih, attn_whh, (PRENETD + FENC + ARNN) / ksplit, ws_gate);
    lstm_pointwise<<<256, 256, 0, stream>>>(ws_gate, ksplit, attn_bih, attn_bhh,
                                            attention_cell, o_ah, o_ac);

    // 3) GMM attention
    gmm_mlp<<<BATCH, 32, 0, stream>>>(o_ah, attention_context, mlp_w, mlp_b, ws_p);
    gmm_params<<<1, BATCH, 0, stream>>>(ws_p, last_mu, ws_wz, ws_sg, o_mu);
    gmm_energy_softmax<<<BATCH, 256, 0, stream>>>(ws_wz, o_mu, ws_sg, mask, o_attn);
    context_partial<<<dim3(FENC / 256, BATCH, tsplit), 256, 0, stream>>>(
        o_attn, memory, tchunk, ws_ctxp);
    context_combine<<<(BATCH * FENC + 255) / 256, 256, 0, stream>>>(ws_ctxp, tsplit, o_ctx);

    // 4) decoder LSTM: [ah(1024) | ctx(512)] + dh(1024), K=2560
    lstm_gates_wmma<<<dim3(32, ksplit), 256, 0, stream>>>(
        o_ah, ARNN, o_ctx, FENC, decoder_hidden, DRNN,
        dec_wih, dec_whh, (ARNN + FENC + DRNN) / ksplit, ws_gate);
    lstm_pointwise<<<256, 256, 0, stream>>>(ws_gate, ksplit, dec_bih, dec_bhh,
                                            decoder_cell, o_dh, o_dc);

    // 5) heads
    heads_kernel<<<BATCH, 96, 0, stream>>>(o_dh, o_ctx, proj_w, proj_b, gate_w, gate_b,
                                           o_decoder, o_gate);
}
